// GNN_80642305950254
// MI455X (gfx1250) — matrix operand order repaired and checked by reading, not compiled, not run on hardware
//
#include <hip/hip_runtime.h>

#define NN 100000   // nodes
#define NE 160000   // edges
#define NB 4096     // batch
#define FN 133
#define FE 14
#define H  512
#define DEPTH_L 5
#define KINIT 160   // 147 padded to mult of 32
#define KE2N  672   // 645 padded to mult of 32
#define CH_GRP 16   // K-chunks staged in LDS at a time (16*4KB = 64KB)

typedef __attribute__((ext_vector_type(16))) __bf16 v16bf;
typedef __attribute__((ext_vector_type(8)))  float  v8f;
typedef __attribute__((ext_vector_type(4)))  unsigned u32x4;
typedef __attribute__((ext_vector_type(4)))  float  f32x4;

union Frag { u32x4 q[2]; v16bf v; };

__device__ __forceinline__ unsigned short f2bf(float f) {
  unsigned u = __float_as_uint(f);
  return (unsigned short)((u + 0x7fffu + ((u >> 16) & 1u)) >> 16);  // RNE
}

__device__ __forceinline__ void atom_add(float* p, float v) {
  __hip_atomic_fetch_add(p, v, __ATOMIC_RELAXED, __HIP_MEMORY_SCOPE_AGENT);
}

// ---------------------------------------------------------------------------
// GEMM: Out[M,512] = epilogue( A[M,K](bf16) @ Wp[K,512](packed bf16) )
// Block: 4 waves, tile M=128 x N=64. Each wave: 32(M) x 64(N) -> 8 accumulators.
// B panel (64 cols, CH_GRP K-chunks) staged in LDS once per group, shared by all
// 4 waves. Per chunk per wave: 4 global b128 (A) + 8 ds b128 (B) + 8 WMMAs.
// ---------------------------------------------------------------------------
__global__ __launch_bounds__(128)
void gemm_bf16(const unsigned short* __restrict__ A, const unsigned* __restrict__ Wp,
               const float* __restrict__ bias, const float* __restrict__ resid,
               float* __restrict__ Out, int M, int K, int do_relu) {
  __shared__ u32x4 sB[CH_GRP * 256];   // 64 KB: [chunk][hi][n_local] 32B pairs

  const int wave = threadIdx.x >> 5;
  const int lane = threadIdx.x & 31;
  const int l    = lane & 15;
  const int hi   = lane >> 4;
  const int m0   = (int)blockIdx.x * 128 + wave * 32;
  const int n0   = (int)blockIdx.y * 64;

  // clamp A rows (keeps EXEC all-ones for WMMA; stores are guarded below)
  int r0 = m0 + l;      if (r0 > M - 1) r0 = M - 1;
  int r1 = m0 + 16 + l; if (r1 > M - 1) r1 = M - 1;
  const unsigned short* arow0 = A + (size_t)r0 * (size_t)K;
  const unsigned short* arow1 = A + (size_t)r1 * (size_t)K;

  const u32x4* Wp4 = (const u32x4*)Wp;
  v8f acc[2][4] = {};

  const int nch = K >> 5;
  for (int g0 = 0; g0 < nch; g0 += CH_GRP) {
    const int gn = (nch - g0 < CH_GRP) ? (nch - g0) : CH_GRP;
    __syncthreads();   // previous group fully consumed before overwrite
    for (int i = threadIdx.x; i < gn * 256; i += 128) {
      const int cl  = i >> 8;          // local chunk
      const int rem = i & 255;
      const int hii = rem >> 7;
      const int nl  = (rem >> 1) & 63;
      const int r4  = rem & 1;
      sB[i] = Wp4[((size_t)(g0 + cl) * H + (size_t)(n0 + nl)) * 4 + 2 * hii + r4];
    }
    __syncthreads();

    for (int cc = 0; cc < gn; ++cc) {
      const int k0 = (g0 + cc) << 5;
      Frag a0, a1;
      a0.q[0] = *(const u32x4*)(arow0 + k0 + hi * 8);        // K = k0+hi*8   .. +7
      a0.q[1] = *(const u32x4*)(arow0 + k0 + 16 + hi * 8);   // K = k0+16+hi*8.. +7
      a1.q[0] = *(const u32x4*)(arow1 + k0 + hi * 8);
      a1.q[1] = *(const u32x4*)(arow1 + k0 + 16 + hi * 8);

      const u32x4* sb = sB + cc * 256;
      Frag bf[4];
#pragma unroll
      for (int t = 0; t < 4; ++t) {
        const int nl = t * 16 + l;
        bf[t].q[0] = sb[hi * 128 + nl * 2];
        bf[t].q[1] = sb[hi * 128 + nl * 2 + 1];
      }
#pragma unroll
      for (int t = 0; t < 4; ++t) {
        acc[0][t] = __builtin_amdgcn_wmma_f32_16x16x32_bf16(
            false, a0.v, false, bf[t].v, (short)0, acc[0][t], false, false);
        acc[1][t] = __builtin_amdgcn_wmma_f32_16x16x32_bf16(
            false, a1.v, false, bf[t].v, (short)0, acc[1][t], false, false);
      }
    }
  }

  // C/D layout: VGPR r, lanes 0-15 -> M=r, N=lane ; lanes 16-31 -> M=r+8, N=lane-16
#pragma unroll
  for (int mh = 0; mh < 2; ++mh) {
#pragma unroll
    for (int t = 0; t < 4; ++t) {
      const int colN = n0 + t * 16 + l;
      const float b = bias[colN];
#pragma unroll
      for (int r = 0; r < 8; ++r) {
        const int rowv = m0 + mh * 16 + hi * 8 + r;
        if (rowv < M) {
          float v = acc[mh][t][r] + b;
          if (resid) v += resid[(size_t)rowv * H + colN];
          if (do_relu) v = fmaxf(v, 0.f);
          Out[(size_t)rowv * H + colN] = v;
        }
      }
    }
  }
}

// Pack f32 weights [Kact,512] (row-major) -> fragment-native bf16 pairs.
// uint index u = ((c*512+n)*2+hi)*8+r  holds bf16(W[c*32+(r>>2)*16+hi*8+2*(r&3)][n]) | bf16(..+1)<<16
__global__ void pack_w(const float* __restrict__ W, unsigned* __restrict__ Wp,
                       int Kact, long long total) {
  long long i = (long long)blockIdx.x * blockDim.x + threadIdx.x;
  if (i >= total) return;
  int r  = (int)(i & 7);
  int hi = (int)((i >> 3) & 1);
  int n  = (int)((i >> 4) & 511);
  int c  = (int)(i >> 13);
  int k  = c * 32 + (r >> 2) * 16 + hi * 8 + (r & 3) * 2;
  float f0 = (k     < Kact) ? W[(size_t)k * H + n]       : 0.f;
  float f1 = (k + 1 < Kact) ? W[(size_t)(k + 1) * H + n] : 0.f;
  Wp[i] = (unsigned)f2bf(f0) | ((unsigned)f2bf(f1) << 16);
}

// xe[e][0:133]=x[row[e]], [133:147]=edge_attr[e], [147:160]=0   (bf16)
__global__ void build_xe(const float* __restrict__ x, const float* __restrict__ ea,
                         const int* __restrict__ row, unsigned short* __restrict__ xe) {
  long long i = (long long)blockIdx.x * blockDim.x + threadIdx.x;
  if (i >= (long long)NE * KINIT) return;
  int e = (int)(i / KINIT);
  int j = (int)(i - (long long)e * KINIT);
  float v = 0.f;
  if (j < FN)            v = x[(size_t)row[e] * FN + j];
  else if (j < FN + FE)  v = ea[(size_t)e * FE + (j - FN)];
  xe[i] = f2bf(v);
}

// dst[idx[e]] += src[e]   over [R,512] rows (segment_sum via HW f32 atomics)
__global__ void scatter_add(const float* __restrict__ src, const int* __restrict__ idx,
                            float* __restrict__ dst, long long total) {
  long long i = (long long)blockIdx.x * blockDim.x + threadIdx.x;
  if (i >= total) return;
  int e = (int)(i >> 9);
  int k = (int)(i & 511);
  atom_add(&dst[(size_t)idx[e] * H + k], src[i]);
}

// msg[e][k] = bf16( a[row[e]][k] - h[e^1][k] )
__global__ void build_msg(const float* __restrict__ a, const float* __restrict__ h,
                          const int* __restrict__ row, unsigned short* __restrict__ msg) {
  long long i = (long long)blockIdx.x * blockDim.x + threadIdx.x;
  if (i >= (long long)NE * H) return;
  int e = (int)(i >> 9);
  int k = (int)(i & 511);
  float v = a[(size_t)row[e] * H + k] - h[(size_t)(e ^ 1) * H + k];
  msg[i] = f2bf(v);
}

// q[n][0:133]=x[n], [133:645]=s_node[n], [645:672]=0   (bf16)
__global__ void build_q(const float* __restrict__ x, const float* __restrict__ a,
                        unsigned short* __restrict__ q) {
  long long i = (long long)blockIdx.x * blockDim.x + threadIdx.x;
  if (i >= (long long)NN * KE2N) return;
  int n = (int)(i / KE2N);
  int j = (int)(i - (long long)n * KE2N);
  float v = 0.f;
  if (j < FN)           v = x[(size_t)n * FN + j];
  else if (j < FN + H)  v = a[(size_t)n * H + (j - FN)];
  q[i] = f2bf(v);
}

__global__ void zero_f32(float* __restrict__ p, long long n) {
  long long i = (long long)blockIdx.x * blockDim.x + threadIdx.x;
  if (i < n) p[i] = 0.f;
}

__global__ void f32_to_bf16(const float* __restrict__ s, unsigned short* __restrict__ d,
                            long long n) {
  long long i = (long long)blockIdx.x * blockDim.x + threadIdx.x;
  if (i < n) d[i] = f2bf(s[i]);
}

// out[b] = t[b,:] . W_ffn2 + b_ffn2   (t already relu'd by GEMM epilogue)
__global__ void final_dot(const float* __restrict__ t, const float* __restrict__ w2,
                          const float* __restrict__ b2, float* __restrict__ out) {
  int b = blockIdx.x * blockDim.x + threadIdx.x;
  if (b >= NB) return;
  const f32x4* tv = (const f32x4*)(t + (size_t)b * H);
  const f32x4* wv = (const f32x4*)w2;
  float s = 0.f;
  for (int k = 0; k < H / 4; ++k) {
    f32x4 av = tv[k], ww = wv[k];
    s += av.x * ww.x + av.y * ww.y + av.z * ww.z + av.w * ww.w;
  }
  out[b] = s + b2[0];
}

// ---------------------------------------------------------------------------
extern "C" void kernel_launch(void* const* d_in, const int* in_sizes, int n_in,
                              void* d_out, int out_size, void* d_ws, size_t ws_size,
                              hipStream_t stream) {
  (void)in_sizes; (void)n_in; (void)out_size; (void)ws_size;

  const float* x         = (const float*)d_in[0];
  const float* edge_attr = (const float*)d_in[1];
  const int*   edge_idx  = (const int*)  d_in[2];
  const int*   batch     = (const int*)  d_in[3];
  const float* W_init    = (const float*)d_in[4];
  const float* b_init    = (const float*)d_in[5];
  const float* W_conv    = (const float*)d_in[6];
  const float* b_conv    = (const float*)d_in[7];
  const float* W_e2n     = (const float*)d_in[8];
  const float* b_e2n     = (const float*)d_in[9];
  const float* W_ffn1    = (const float*)d_in[10];
  const float* b_ffn1    = (const float*)d_in[11];
  const float* W_ffn2    = (const float*)d_in[12];
  const float* b_ffn2    = (const float*)d_in[13];
  float* out = (float*)d_out;

  const int* row = edge_idx;          // edge_index[0]
  const int* col = edge_idx + NE;     // edge_index[1]

  // ---- workspace carve-out ----
  char* ws = (char*)d_ws;
  size_t off = 0;
  auto alloc = [&](size_t bytes) -> void* {
    off = (off + 255) & ~(size_t)255;
    void* p = ws + off; off += bytes; return p;
  };
  float*          h0     = (float*)alloc((size_t)NE * H * 4);
  float*          h      = (float*)alloc((size_t)NE * H * 4);          // also holds hn
  float*          a      = (float*)alloc((size_t)NN * H * 4);          // agg / s_node
  unsigned short* msg    = (unsigned short*)alloc((size_t)NE * H * 2); // also xe / q / pooled_bf16
  float*          pooled = (float*)alloc((size_t)NB * H * 4);
  float*          tbuf   = (float*)alloc((size_t)NB * H * 4);
  unsigned* Wp_init = (unsigned*)alloc((size_t)KINIT * H * 2);
  unsigned* Wp_conv = (unsigned*)alloc((size_t)DEPTH_L * H * H * 2);
  unsigned* Wp_e2n  = (unsigned*)alloc((size_t)KE2N * H * 2);
  unsigned* Wp_ffn1 = (unsigned*)alloc((size_t)H * H * 2);

  auto blocks = [](long long n) { return (unsigned)((n + 255) / 256); };

  // ---- pack all weights to fragment-native bf16 (tiny, once per launch) ----
  const long long tInit = (long long)(KINIT / 32) * H * 16;
  const long long tConv = (long long)(H / 32) * H * 16;
  const long long tE2n  = (long long)(KE2N / 32) * H * 16;
  pack_w<<<blocks(tInit), 256, 0, stream>>>(W_init, Wp_init, FN + FE, tInit);
  for (int l = 0; l < DEPTH_L; ++l)
    pack_w<<<blocks(tConv), 256, 0, stream>>>(W_conv + (size_t)l * H * H,
                                              Wp_conv + (size_t)l * (H * H / 2), H, tConv);
  pack_w<<<blocks(tE2n), 256, 0, stream>>>(W_e2n, Wp_e2n, FN + H, tE2n);
  pack_w<<<blocks(tConv), 256, 0, stream>>>(W_ffn1, Wp_ffn1, H, tConv);

  // ---- h0 = relu([x[row]|edge_attr] @ W_init + b_init) ----
  build_xe<<<blocks((long long)NE * KINIT), 256, 0, stream>>>(x, edge_attr, row, msg);
  dim3 gE((NE + 127) / 128, H / 64);
  gemm_bf16<<<gE, 128, 0, stream>>>(msg, Wp_init, b_init, nullptr, h0, NE, KINIT, 1);

  // ---- DEPTH message-passing layers ----
  const long long nA  = (long long)NN * H;
  const long long nEH = (long long)NE * H;
  const float* h_in = h0;
  for (int l = 0; l < DEPTH_L; ++l) {
    zero_f32   <<<blocks(nA),  256, 0, stream>>>(a, nA);
    scatter_add<<<blocks(nEH), 256, 0, stream>>>(h_in, col, a, nEH);      // a = seg_sum(h, col)
    build_msg  <<<blocks(nEH), 256, 0, stream>>>(a, h_in, row, msg);      // a[row]-h[rev]
    gemm_bf16  <<<gE, 128, 0, stream>>>(msg, Wp_conv + (size_t)l * (H * H / 2),
                                        b_conv + (size_t)l * H, h0, h, NE, H, 1);
    h_in = h;
  }

  // ---- s_node = seg_sum(h, col) ; hn = relu([x|s_node] @ W_e2n + b) ----
  zero_f32   <<<blocks(nA),  256, 0, stream>>>(a, nA);
  scatter_add<<<blocks(nEH), 256, 0, stream>>>(h_in, col, a, nEH);
  build_q    <<<blocks((long long)NN * KE2N), 256, 0, stream>>>(x, a, msg);
  dim3 gN((NN + 127) / 128, H / 64);
  gemm_bf16<<<gN, 128, 0, stream>>>(msg, Wp_e2n, b_e2n, nullptr, h, NN, KE2N, 1); // hn -> h

  // ---- pooled = seg_sum(hn, batch) ; FFN ----
  const long long nP = (long long)NB * H;
  zero_f32   <<<blocks(nP), 256, 0, stream>>>(pooled, nP);
  scatter_add<<<blocks((long long)NN * H), 256, 0, stream>>>(h, batch, pooled, (long long)NN * H);
  f32_to_bf16<<<blocks(nP), 256, 0, stream>>>(pooled, msg, nP);
  dim3 gB((NB + 127) / 128, H / 64);
  gemm_bf16<<<gB, 128, 0, stream>>>(msg, Wp_ffn1, b_ffn1, nullptr, tbuf, NB, H, 1);
  final_dot<<<NB / 256, 256, 0, stream>>>(tbuf, W_ffn2, b_ffn2, out);
}
